// Transformer_P_13477607375003
// MI455X (gfx1250) — matrix operand order repaired and checked by reading, not compiled, not run on hardware
//
#include <hip/hip_runtime.h>

// ---------------- problem constants ----------------
#define D_MODEL 512
#define HEADS   8
#define DH      64
#define N0      22400          // input tokens
#define N1      22501          // + cls
#define NSQ     22500          // 150*150
#define HW      150
#define NA      22528          // padded seq (88*256)
#define PADF    27             // front zero pad
#define LM      256            // landmarks (M)
#define RESK    33

typedef __attribute__((ext_vector_type(16))) __bf16 bf16x16;
typedef __attribute__((ext_vector_type(8)))  float  f32x8;

// ---------------- CDNA5 async-to-LDS helpers ----------------
// GV-mode async copy: 16B global -> LDS, tracked by ASYNCcnt (no VGPR staging).
__device__ __forceinline__ void async_ld_b128(const __bf16* g, __bf16* l)
{
    unsigned laddr = (unsigned)(unsigned long long)l;   // LDS aperture: low 32 bits = LDS byte addr
    asm volatile("global_load_async_to_lds_b128 %0, %1, off"
                 :: "v"(laddr), "v"(g) : "memory");
}
__device__ __forceinline__ void wait_async0() { asm volatile("s_wait_asynccnt 0x0" ::: "memory"); }
__device__ __forceinline__ void wait_async2() { asm volatile("s_wait_asynccnt 0x2" ::: "memory"); }
__device__ __forceinline__ void wait_async3() { asm volatile("s_wait_asynccnt 0x3" ::: "memory"); }

// ---------------- WMMA GEMM ----------------
// C[M x N] = alpha * A[M x K] * op(B) (+ bias), bf16 in, f32 accum.
// transB=0: B is KxN row-major; transB=1: op(B)=Bmat^T with Bmat NxK row-major.
// Batched via sA/sB/sC element strides; split-K accumulates into f32 C with atomics.
// Double-buffered LDS; A tile (and B tile when transB=1) streamed with
// global_load_async_to_lds_b128 so next-tile loads overlap current-tile WMMAs.
#define BM 128
#define BN 64
#define BKK 32
#define AST 40   // LDS row pitch (bf16 elems) -> 80B rows, 16B aligned vectors
#define BST 40

__global__ __launch_bounds__(256)
void gemm_bf16_wmma(const __bf16* __restrict__ A, const __bf16* __restrict__ B,
                    float* __restrict__ Cf, __bf16* __restrict__ Cb,
                    const float* __restrict__ bias,
                    int Mrows, int Ncols, int K,
                    int lda, int ldb, int ldc,
                    int transB, float alpha,
                    long sA, long sB, long sC,
                    int splitK, int kChunk)
{
    __shared__ __align__(16) __bf16 As[2][BM * AST];
    __shared__ __align__(16) __bf16 Bs[2][BN * BST];
    __shared__ __align__(16) __bf16 dummy[8];          // sink for masked async lanes

    int bz    = blockIdx.z;
    int batch = bz / splitK;
    int ks    = bz - batch * splitK;
    int kBeg  = ks * kChunk;
    int kEnd  = kBeg + kChunk; if (kEnd > K) kEnd = K;
    int nsteps = (kEnd > kBeg) ? (kEnd - kBeg) / BKK : 0;   // K always a multiple of 32

    const __bf16* Ab = A + (long)batch * sA;
    const __bf16* Bb = B + (long)batch * sB;
    long cofs = (long)batch * sC;

    int tileM = blockIdx.y * BM, tileN = blockIdx.x * BN;
    int tid = threadIdx.x, lane = tid & 31, wid = tid >> 5;
    int wm = (wid >> 1) * 32, wn = (wid & 1) * 32;   // 4x2 wave grid
    int lhalf = lane >> 4, lrow = lane & 15;

    // M-edge blocks: pre-zero both A buffers once; OOB rows are OOB for every
    // k-step and their async loads are redirected to `dummy`, so they stay 0.
    if (tileM + BM > Mrows) {
        int4 z = make_int4(0, 0, 0, 0);
        for (int i = tid; i < (BM * AST) / 8; i += 256) {
            ((int4*)As[0])[i] = z;
            ((int4*)As[1])[i] = z;
        }
    }
    __syncthreads();

    // per-thread tile-load descriptors
    int ar0 = (tid) >> 2,        aq0 = (tid & 3) * 8;         // A row/quarter, part 0
    int ar1 = (tid + 256) >> 2,  aq1 = ((tid + 256) & 3) * 8; // part 1
    int bn1 = tid >> 2,          bq1 = (tid & 3) * 8;         // B transB=1
    int bk0 = tid >> 3,          bg0 = (tid & 7) * 8;         // B transB=0

    auto issueA = [&](int buf, int k0) {
        int gr0 = tileM + ar0, gr1 = tileM + ar1;
        const __bf16* s0 = (gr0 < Mrows) ? (Ab + (long)gr0 * lda + k0 + aq0) : (Ab + k0 + aq0);
        __bf16*       d0 = (gr0 < Mrows) ? &As[buf][ar0 * AST + aq0] : dummy;
        const __bf16* s1 = (gr1 < Mrows) ? (Ab + (long)gr1 * lda + k0 + aq1) : (Ab + k0 + aq1);
        __bf16*       d1 = (gr1 < Mrows) ? &As[buf][ar1 * AST + aq1] : dummy;
        async_ld_b128(s0, d0);
        async_ld_b128(s1, d1);
    };
    auto issueB = [&](int buf, int k0) {
        if (transB) {
            async_ld_b128(Bb + (long)(tileN + bn1) * ldb + k0 + bq1,
                          &Bs[buf][bn1 * BST + bq1]);
        } else {   // register transpose into Bt[n][k] (synchronous)
            int4 v = *(const int4*)(Bb + (long)(k0 + bk0) * ldb + tileN + bg0);
            const __bf16* pv = (const __bf16*)&v;
            #pragma unroll
            for (int j = 0; j < 8; ++j) Bs[buf][(bg0 + j) * BST + bk0] = pv[j];
        }
    };

    f32x8 acc[2][2] = {};

    if (nsteps > 0) { issueA(0, kBeg); issueB(0, kBeg); }

    for (int s = 0; s < nsteps; ++s) {
        int k0  = kBeg + s * BKK;
        int cur = s & 1, nxt = cur ^ 1;
        bool hasNext = (s + 1 < nsteps);
        if (hasNext) { issueA(nxt, k0 + BKK); issueB(nxt, k0 + BKK); }
        // release current tile: async ops complete in order, so waiting down to
        // the count just issued for the next tile means the current one landed.
        if (hasNext) { if (transB) wait_async3(); else wait_async2(); }
        else         wait_async0();
        __syncthreads();

        union Frag { bf16x16 v; int u[8]; };
        Frag af[2], bfr[2];
        #pragma unroll
        for (int f = 0; f < 2; ++f) {
            int arr = wm + f * 16 + lrow;  // A row
            int bnn = wn + f * 16 + lrow;  // B col
            #pragma unroll
            for (int p = 0; p < 8; ++p) {
                int ka = ((p < 4) ? 0 : 16) + lhalf * 8 + 2 * (p & 3); // A 16x32 layout
                int kb = ((p < 4) ? 0 : 16) + lhalf * 8 + 2 * (p & 3); // B 32x16 layout
                af[f].u[p]  = *(const int*)(&As[cur][arr * AST + ka]);
                bfr[f].u[p] = *(const int*)(&Bs[cur][bnn * BST + kb]);
            }
        }
        #pragma unroll
        for (int i = 0; i < 2; ++i)
            #pragma unroll
            for (int j = 0; j < 2; ++j)
                acc[i][j] = __builtin_amdgcn_wmma_f32_16x16x32_bf16(
                    false, af[i].v, false, bfr[j].v, (short)0, acc[i][j], false, false);
        __syncthreads();   // all waves done with `cur` -> next iter may refill it
    }

    // C layout: VGPR r -> row r + 8*lhalf, col = lrow
    #pragma unroll
    for (int i = 0; i < 2; ++i)
        #pragma unroll
        for (int j = 0; j < 2; ++j)
            #pragma unroll
            for (int r = 0; r < 8; ++r) {
                int row = tileM + wm + i * 16 + r + lhalf * 8;
                int col = tileN + wn + j * 16 + lrow;
                if (row < Mrows && col < Ncols) {
                    float v = acc[i][j][r] * alpha;
                    long off = cofs + (long)row * ldc + col;
                    if (splitK > 1) {
                        atomicAdd(Cf + off, v);
                    } else if (Cf) {
                        if (bias) v += bias[col];
                        Cf[off] = v;
                    } else {
                        Cb[off] = (__bf16)v;
                    }
                }
            }
}

// ---------------- support kernels ----------------
__global__ __launch_bounds__(256)
void build_h_k(const float* __restrict__ feats, const float* __restrict__ cls,
               float* __restrict__ h)
{
    long idx = (long)blockIdx.x * 256 + threadIdx.x;
    if (idx >= (long)N1 * D_MODEL) return;
    int c = (int)(idx & (D_MODEL - 1));
    long r = idx >> 9;
    float v;
    if (r == 0) v = cls[c];
    else if (r <= N0) v = feats[(r - 1) * D_MODEL + c];
    else v = feats[(r - 1 - N0) * D_MODEL + c];   // wrap-around rows
    h[idx] = v;
}

__global__ __launch_bounds__(256)
void layernorm_pad_k(const float* __restrict__ h, const float* __restrict__ g,
                     const float* __restrict__ b, __bf16* __restrict__ out)
{
    int row = blockIdx.x;      // 0..NA-1
    int tid = threadIdx.x;
    __shared__ float red[256];
    __bf16* o = out + (long)row * D_MODEL;
    if (row < PADF) { o[tid] = (__bf16)0.f; o[tid + 256] = (__bf16)0.f; return; }
    const float* x = h + (long)(row - PADF) * D_MODEL;
    float v0 = x[tid], v1 = x[tid + 256];
    red[tid] = v0 + v1; __syncthreads();
    for (int s = 128; s > 0; s >>= 1) { if (tid < s) red[tid] += red[tid + s]; __syncthreads(); }
    float mu = red[0] * (1.f / D_MODEL); __syncthreads();
    float d0 = v0 - mu, d1 = v1 - mu;
    red[tid] = d0 * d0 + d1 * d1; __syncthreads();
    for (int s = 128; s > 0; s >>= 1) { if (tid < s) red[tid] += red[tid + s]; __syncthreads(); }
    float rs = rsqrtf(red[0] * (1.f / D_MODEL) + 1e-5f);
    o[tid]       = (__bf16)(d0 * rs * g[tid] + b[tid]);
    o[tid + 256] = (__bf16)(d1 * rs * g[tid + 256] + b[tid + 256]);
}

__global__ __launch_bounds__(256)
void convert_wqkv_k(const float* __restrict__ w, __bf16* __restrict__ o)
{   // fold q *= DH^-0.5 (=0.125 exactly) into first 512 output columns
    long i = (long)blockIdx.x * 256 + threadIdx.x;
    if (i >= (long)D_MODEL * 3 * D_MODEL) return;
    int col = (int)(i % (3 * D_MODEL));
    float s = (col < D_MODEL) ? 0.125f : 1.f;
    o[i] = (__bf16)(w[i] * s);
}

__global__ __launch_bounds__(256)
void f32_to_bf16_k(const float* __restrict__ s, __bf16* __restrict__ d, long n)
{ long i = (long)blockIdx.x * 256 + threadIdx.x; if (i < n) d[i] = (__bf16)s[i]; }

__global__ __launch_bounds__(256)
void copy_f32_k(const float* __restrict__ s, float* __restrict__ d, long n)
{ long i = (long)blockIdx.x * 256 + threadIdx.x; if (i < n) d[i] = s[i]; }

__global__ __launch_bounds__(256)
void add_inplace_k(float* __restrict__ d, const float* __restrict__ s, long n)
{ long i = (long)blockIdx.x * 256 + threadIdx.x; if (i < n) d[i] += s[i]; }

__global__ __launch_bounds__(256)
void zero_f32_k(float* __restrict__ d, long n)
{ long i = (long)blockIdx.x * 256 + threadIdx.x; if (i < n) d[i] = 0.f; }

__global__ void zero_u32_k(unsigned* d, int n)
{ int i = threadIdx.x; if (i < n) d[i] = 0u; }

__global__ __launch_bounds__(256)
void landmark_k(const __bf16* __restrict__ qkv, __bf16* __restrict__ ql,
                __bf16* __restrict__ kl)
{   // landmark i = mean of rows [i*88, (i+1)*88)
    int i = blockIdx.x, tid = threadIdx.x;
    const int l = NA / LM;
    float sq0 = 0, sq1 = 0, sk0 = 0, sk1 = 0;
    const __bf16* base = qkv + (long)i * l * (3 * D_MODEL);
    for (int j = 0; j < l; ++j) {
        const __bf16* r = base + (long)j * (3 * D_MODEL);
        sq0 += (float)r[tid];            sq1 += (float)r[tid + 256];
        sk0 += (float)r[D_MODEL + tid];  sk1 += (float)r[D_MODEL + tid + 256];
    }
    float inv = 1.f / l;
    ql[(long)i * D_MODEL + tid]       = (__bf16)(sq0 * inv);
    ql[(long)i * D_MODEL + tid + 256] = (__bf16)(sq1 * inv);
    kl[(long)i * D_MODEL + tid]       = (__bf16)(sk0 * inv);
    kl[(long)i * D_MODEL + tid + 256] = (__bf16)(sk1 * inv);
}

__global__ __launch_bounds__(256)
void softmax_rows_k(__bf16* __restrict__ data, int width)
{
    long row = blockIdx.x;
    __bf16* x = data + row * (long)width;
    __shared__ float red[256];
    int tid = threadIdx.x;
    float mx = -1e30f;
    for (int c = tid; c < width; c += 256) mx = fmaxf(mx, (float)x[c]);
    red[tid] = mx; __syncthreads();
    for (int s = 128; s > 0; s >>= 1) { if (tid < s) red[tid] = fmaxf(red[tid], red[tid + s]); __syncthreads(); }
    mx = red[0]; __syncthreads();
    float sum = 0.f;
    for (int c = tid; c < width; c += 256) sum += __expf((float)x[c] - mx);
    red[tid] = sum; __syncthreads();
    for (int s = 128; s > 0; s >>= 1) { if (tid < s) red[tid] += red[tid + s]; __syncthreads(); }
    float inv = 1.f / red[0];
    for (int c = tid; c < width; c += 256) x[c] = (__bf16)(__expf((float)x[c] - mx) * inv);
}

__global__ __launch_bounds__(256)
void colsum_max_k(const __bf16* __restrict__ a2, float* __restrict__ colmax)
{   // softmax rows sum to 1 -> rowsum max = 1; need global max column sum
    int h = blockIdx.x, c = threadIdx.x;
    const __bf16* m = a2 + (long)h * LM * LM;
    float s = 0.f;
    for (int r = 0; r < LM; ++r) s += (float)m[r * LM + c];
    __shared__ float red[256];
    red[c] = s; __syncthreads();
    for (int o = 128; o > 0; o >>= 1) { if (c < o) red[c] = fmaxf(red[c], red[c + o]); __syncthreads(); }
    if (c == 0) atomicMax((unsigned*)colmax, __float_as_uint(red[0]));  // positive floats
}

__global__ __launch_bounds__(256)
void pinv_init_k(const __bf16* __restrict__ a2, const float* __restrict__ colmax,
                 __bf16* __restrict__ z)
{   // z0 = a2^T / colmax
    long idx = (long)blockIdx.x * 256 + threadIdx.x;
    if (idx >= (long)HEADS * LM * LM) return;
    int j = (int)(idx & 255), i = (int)((idx >> 8) & 255), h = (int)(idx >> 16);
    float inv = 1.f / colmax[0];
    z[idx] = (__bf16)((float)a2[((long)h << 16) + (j << 8) + i] * inv);
}

__global__ __launch_bounds__(256)
void diag_sub_k(const __bf16* __restrict__ t, __bf16* __restrict__ u, float cdiag)
{   // u = cdiag*I - t
    long idx = (long)blockIdx.x * 256 + threadIdx.x;
    if (idx >= (long)HEADS * LM * LM) return;
    int j = (int)(idx & 255), i = (int)((idx >> 8) & 255);
    float v = -(float)t[idx];
    if (i == j) v += cdiag;
    u[idx] = (__bf16)v;
}

__global__ __launch_bounds__(256)
void dwconv_add_k(float* __restrict__ attn, const __bf16* __restrict__ qkv,
                  const float* __restrict__ resw)
{   // attn[n, h*64+d] += sum_t resw[h,t] * v[n+t-16, h*64+d]
    long idx = (long)blockIdx.x * 256 + threadIdx.x;
    if (idx >= (long)NA * D_MODEL) return;
    int c = (int)(idx & (D_MODEL - 1));
    long n = idx >> 9;
    int head = c >> 6;
    float s = 0.f;
    #pragma unroll
    for (int t = 0; t < RESK; ++t) {
        long m = n + t - RESK / 2;
        if (m >= 0 && m < NA)
            s += resw[head * RESK + t] * (float)qkv[m * (3 * D_MODEL) + 2 * D_MODEL + c];
    }
    attn[idx] += s;
}

__global__ __launch_bounds__(256)
void ppeg_k(const float* __restrict__ h,
            const float* __restrict__ w7, const float* __restrict__ b7,
            const float* __restrict__ w5, const float* __restrict__ b5,
            const float* __restrict__ w3, const float* __restrict__ b3,
            float* __restrict__ y)
{
    long idx = (long)blockIdx.x * 256 + threadIdx.x;
    if (idx >= (long)NSQ * D_MODEL) return;
    int ch = (int)(idx & 511);
    long p = idx >> 9;
    int r = (int)(p / HW), c = (int)(p % HW);
    float acc = h[(1 + p) * D_MODEL + ch] + b7[ch] + b5[ch] + b3[ch];
    for (int ky = 0; ky < 7; ++ky) { int rr = r + ky - 3; if (rr < 0 || rr >= HW) continue;
        for (int kx = 0; kx < 7; ++kx) { int cc = c + kx - 3; if (cc < 0 || cc >= HW) continue;
            acc += w7[ch * 49 + ky * 7 + kx] * h[(1 + (long)rr * HW + cc) * D_MODEL + ch]; } }
    for (int ky = 0; ky < 5; ++ky) { int rr = r + ky - 2; if (rr < 0 || rr >= HW) continue;
        for (int kx = 0; kx < 5; ++kx) { int cc = c + kx - 2; if (cc < 0 || cc >= HW) continue;
            acc += w5[ch * 25 + ky * 5 + kx] * h[(1 + (long)rr * HW + cc) * D_MODEL + ch]; } }
    for (int ky = 0; ky < 3; ++ky) { int rr = r + ky - 1; if (rr < 0 || rr >= HW) continue;
        for (int kx = 0; kx < 3; ++kx) { int cc = c + kx - 1; if (cc < 0 || cc >= HW) continue;
            acc += w3[ch * 9 + ky * 3 + kx] * h[(1 + (long)rr * HW + cc) * D_MODEL + ch]; } }
    y[idx] = acc;
}

// ---------------- host orchestration ----------------
struct Ws {
    float *h, *tmp, *attn, *a3v_f, *colmax;
    __bf16 *x_bf, *qkv_bf, *ql_bf, *kl_bf, *a1_bf, *a3_bf, *a2_bf;
    __bf16 *z_bf, *z2_bf, *xz_bf, *u_bf, *t_bf, *a3v_bf, *w2_bf;
    __bf16 *wqkv_bf, *wout_bf;
};

static void run_gemm(const void* A, const void* B, float* Cf, void* Cb, const float* bias,
                     int M, int N, int K, int lda, int ldb, int ldc, int transB, float alpha,
                     long sA, long sB, long sC, int batch, int splitK, hipStream_t s)
{
    int kChunk = K;
    if (splitK > 1) { int steps = (K / BKK + splitK - 1) / splitK; kChunk = steps * BKK; }
    dim3 grid((N + BN - 1) / BN, (M + BM - 1) / BM, batch * splitK);
    gemm_bf16_wmma<<<grid, 256, 0, s>>>((const __bf16*)A, (const __bf16*)B, Cf, (__bf16*)Cb,
                                        bias, M, N, K, lda, ldb, ldc, transB, alpha,
                                        sA, sB, sC, splitK, kChunk);
}

static inline int nb(long n) { return (int)((n + 255) / 256); }

static void nystrom_layer(const Ws& w, hipStream_t s,
                          const float* lng, const float* lnb, const float* wqkv,
                          const float* wout, const float* bout, const float* resw)
{
    const long e = (long)LM * LM;
    layernorm_pad_k<<<NA, 256, 0, s>>>(w.h, lng, lnb, w.x_bf);
    convert_wqkv_k<<<nb((long)D_MODEL * 3 * D_MODEL), 256, 0, s>>>(wqkv, w.wqkv_bf);
    f32_to_bf16_k<<<nb((long)D_MODEL * D_MODEL), 256, 0, s>>>(wout, w.wout_bf, (long)D_MODEL * D_MODEL);

    // qkv = x @ Wqkv   (q pre-scaled via weights)
    run_gemm(w.x_bf, w.wqkv_bf, nullptr, w.qkv_bf, nullptr,
             NA, 3 * D_MODEL, D_MODEL, D_MODEL, 3 * D_MODEL, 3 * D_MODEL, 0, 1.f, 0, 0, 0, 1, 1, s);
    landmark_k<<<LM, 256, 0, s>>>(w.qkv_bf, w.ql_bf, w.kl_bf);

    // a2 = softmax(q_l k_l^T)
    run_gemm(w.ql_bf, w.kl_bf, nullptr, w.a2_bf, nullptr,
             LM, LM, DH, D_MODEL, D_MODEL, LM, 1, 1.f, DH, DH, e, HEADS, 1, s);
    softmax_rows_k<<<HEADS * LM, 256, 0, s>>>(w.a2_bf, LM);
    // a1 = softmax(q k_l^T)
    run_gemm(w.qkv_bf, w.kl_bf, nullptr, w.a1_bf, nullptr,
             NA, LM, DH, 3 * D_MODEL, D_MODEL, LM, 1, 1.f, DH, DH, (long)NA * LM, HEADS, 1, s);
    softmax_rows_k<<<HEADS * NA, 256, 0, s>>>(w.a1_bf, LM);
    // a3 = softmax(q_l k^T)
    run_gemm(w.ql_bf, w.qkv_bf + D_MODEL, nullptr, w.a3_bf, nullptr,
             LM, NA, DH, D_MODEL, 3 * D_MODEL, NA, 1, 1.f, DH, DH, (long)LM * NA, HEADS, 1, s);
    softmax_rows_k<<<HEADS * LM, 256, 0, s>>>(w.a3_bf, NA);

    // a3v = a3 @ v  (split-K: K=22528, otherwise only 16 blocks)
    zero_f32_k<<<nb((long)HEADS * LM * DH), 256, 0, s>>>(w.a3v_f, (long)HEADS * LM * DH);
    run_gemm(w.a3_bf, w.qkv_bf + 2 * D_MODEL, w.a3v_f, nullptr, nullptr,
             LM, DH, NA, NA, 3 * D_MODEL, DH, 0, 1.f, (long)LM * NA, DH, (long)LM * DH, HEADS, 32, s);
    f32_to_bf16_k<<<nb((long)HEADS * LM * DH), 256, 0, s>>>(w.a3v_f, w.a3v_bf, (long)HEADS * LM * DH);

    // pinv iteration (Newton–Schulz, batched 256^3 WMMA GEMMs)
    zero_u32_k<<<1, 32, 0, s>>>((unsigned*)w.colmax, 1);
    colsum_max_k<<<HEADS, 256, 0, s>>>(w.a2_bf, w.colmax);
    pinv_init_k<<<nb((long)HEADS * e), 256, 0, s>>>(w.a2_bf, w.colmax, w.z_bf);
    __bf16* zc = w.z_bf; __bf16* zn = w.z2_bf;
    for (int it = 0; it < 6; ++it) {
        run_gemm(w.a2_bf, zc, nullptr, w.xz_bf, nullptr, LM, LM, LM, LM, LM, LM, 0, 1.f, e, e, e, HEADS, 1, s);
        diag_sub_k<<<nb((long)HEADS * e), 256, 0, s>>>(w.xz_bf, w.u_bf, 7.f);
        run_gemm(w.xz_bf, w.u_bf, nullptr, w.t_bf, nullptr, LM, LM, LM, LM, LM, LM, 0, 1.f, e, e, e, HEADS, 1, s);
        diag_sub_k<<<nb((long)HEADS * e), 256, 0, s>>>(w.t_bf, w.u_bf, 15.f);
        run_gemm(w.xz_bf, w.u_bf, nullptr, w.t_bf, nullptr, LM, LM, LM, LM, LM, LM, 0, 1.f, e, e, e, HEADS, 1, s);
        diag_sub_k<<<nb((long)HEADS * e), 256, 0, s>>>(w.t_bf, w.u_bf, 13.f);
        run_gemm(zc, w.u_bf, nullptr, zn, nullptr, LM, LM, LM, LM, LM, LM, 0, 0.25f, e, e, e, HEADS, 1, s);
        __bf16* t = zc; zc = zn; zn = t;
    }
    // w2 = pinv(a2) @ (a3 v)
    run_gemm(zc, w.a3v_bf, nullptr, w.w2_bf, nullptr,
             LM, DH, LM, LM, DH, DH, 0, 1.f, e, (long)LM * DH, (long)LM * DH, HEADS, 1, s);
    // out = a1 @ w2  -> f32 attn (per-head column block via sC=64)
    run_gemm(w.a1_bf, w.w2_bf, w.attn, nullptr, nullptr,
             NA, DH, LM, LM, DH, D_MODEL, 0, 1.f, (long)NA * LM, (long)LM * DH, DH, HEADS, 1, s);
    // + depthwise 33-tap residual conv of v
    dwconv_add_k<<<nb((long)NA * D_MODEL), 256, 0, s>>>(w.attn, w.qkv_bf, resw);
    // out projection on last N1 rows, then residual into h
    f32_to_bf16_k<<<nb((long)N1 * D_MODEL), 256, 0, s>>>(w.attn + (long)PADF * D_MODEL, w.x_bf, (long)N1 * D_MODEL);
    run_gemm(w.x_bf, w.wout_bf, w.tmp, nullptr, bout,
             N1, D_MODEL, D_MODEL, D_MODEL, D_MODEL, D_MODEL, 0, 1.f, 0, 0, 0, 1, 1, s);
    add_inplace_k<<<nb((long)N1 * D_MODEL), 256, 0, s>>>(w.h, w.tmp, (long)N1 * D_MODEL);
}

extern "C" void kernel_launch(void* const* d_in, const int* in_sizes, int n_in,
                              void* d_out, int out_size, void* d_ws, size_t ws_size,
                              hipStream_t stream)
{
    (void)in_sizes; (void)n_in; (void)out_size; (void)ws_size;
    const float* features = (const float*)d_in[0];
    const float* cls      = (const float*)d_in[1];
    const float* ln1_g = (const float*)d_in[2];
    const float* ln1_b = (const float*)d_in[3];
    const float* qkv1_w = (const float*)d_in[4];
    const float* out1_w = (const float*)d_in[5];
    const float* out1_b = (const float*)d_in[6];
    const float* res1_w = (const float*)d_in[7];
    const float* pe_w7 = (const float*)d_in[8];
    const float* pe_b7 = (const float*)d_in[9];
    const float* pe_w5 = (const float*)d_in[10];
    const float* pe_b5 = (const float*)d_in[11];
    const float* pe_w3 = (const float*)d_in[12];
    const float* pe_b3 = (const float*)d_in[13];
    const float* ln2_g = (const float*)d_in[14];
    const float* ln2_b = (const float*)d_in[15];
    const float* qkv2_w = (const float*)d_in[16];
    const float* out2_w = (const float*)d_in[17];
    const float* out2_b = (const float*)d_in[18];
    const float* res2_w = (const float*)d_in[19];

    char* base = (char*)d_ws;
    size_t off = 0;
    auto take = [&](size_t bytes) -> void* {
        void* p = base + off;
        off = (off + bytes + 255) & ~(size_t)255;
        return p;
    };
    Ws w;
    w.h      = (float*)take((size_t)N1 * D_MODEL * 4);
    w.tmp    = (float*)take((size_t)N1 * D_MODEL * 4);
    w.attn   = (float*)take((size_t)NA * D_MODEL * 4);
    w.a3v_f  = (float*)take((size_t)HEADS * LM * DH * 4);
    w.colmax = (float*)take(256);
    w.x_bf   = (__bf16*)take((size_t)NA * D_MODEL * 2);
    w.qkv_bf = (__bf16*)take((size_t)NA * 3 * D_MODEL * 2);
    w.ql_bf  = (__bf16*)take((size_t)LM * D_MODEL * 2);
    w.kl_bf  = (__bf16*)take((size_t)LM * D_MODEL * 2);
    w.a1_bf  = (__bf16*)take((size_t)HEADS * NA * LM * 2);
    w.a3_bf  = (__bf16*)take((size_t)HEADS * LM * NA * 2);
    w.a2_bf  = (__bf16*)take((size_t)HEADS * LM * LM * 2);
    w.z_bf   = (__bf16*)take((size_t)HEADS * LM * LM * 2);
    w.z2_bf  = (__bf16*)take((size_t)HEADS * LM * LM * 2);
    w.xz_bf  = (__bf16*)take((size_t)HEADS * LM * LM * 2);
    w.u_bf   = (__bf16*)take((size_t)HEADS * LM * LM * 2);
    w.t_bf   = (__bf16*)take((size_t)HEADS * LM * LM * 2);
    w.a3v_bf = (__bf16*)take((size_t)HEADS * LM * DH * 2);
    w.w2_bf  = (__bf16*)take((size_t)HEADS * LM * DH * 2);
    w.wqkv_bf = (__bf16*)take((size_t)D_MODEL * 3 * D_MODEL * 2);
    w.wout_bf = (__bf16*)take((size_t)D_MODEL * D_MODEL * 2);

    // h = [cls ; features ; features[:100]]
    build_h_k<<<nb((long)N1 * D_MODEL), 256, 0, stream>>>(features, cls, w.h);
    // layer 1
    nystrom_layer(w, stream, ln1_g, ln1_b, qkv1_w, out1_w, out1_b, res1_w);
    // PPEG
    ppeg_k<<<nb((long)NSQ * D_MODEL), 256, 0, stream>>>(w.h, pe_w7, pe_b7, pe_w5, pe_b5, pe_w3, pe_b3, w.tmp);
    copy_f32_k<<<nb((long)NSQ * D_MODEL), 256, 0, stream>>>(w.tmp, w.h + D_MODEL, (long)NSQ * D_MODEL);
    // layer 2
    nystrom_layer(w, stream, ln2_g, ln2_b, qkv2_w, out2_w, out2_b, res2_w);
    // output = [h[0] ; h[1:]] == flat h
    copy_f32_k<<<nb((long)N1 * D_MODEL), 256, 0, stream>>>(w.h, (float*)d_out, (long)N1 * D_MODEL);
}
